// StepwiseMonotonicMultiheadAttention_24206435680583
// MI455X (gfx1250) — compile-verified, matrix operand order checked by reading, not run on hardware
//
#include <hip/hip_runtime.h>
#include <hip/hip_bf16.h>

// ---------------------------------------------------------------------------
// Stepwise Monotonic Multihead Attention — CDNA5 (gfx1250) implementation.
// f16 WMMA (v_wmma_f32_16x16x32_f16) for the three big GEMMs, VALU conv,
// LDS-based softmax/reductions/transpose, barrier-stepped monotonic scan
// with LDS tile-transposed output so every WMMA fragment load is b128-able.
// ---------------------------------------------------------------------------

#define B_      16
#define QLEN_   512
#define KLEN_   1024
#define DMODEL_ 384
#define NHEAD_  3
#define DK_     128
#define KW_     13
#define PADW_   6

typedef __attribute__((ext_vector_type(16))) _Float16 v16h;
typedef __attribute__((ext_vector_type(8)))  float    v8f;

// ---- WMMA fragment loader -------------------------------------------------
// Element (i, kslot) lives at base[i*rs + kslot*cs].
// For A fragments: i = M row.  For B fragments: i = N column.
// Lane mapping (ISA 7.12.2, 16-bit 16x32): lane&15 -> i, halves j:
//   j in 0..7  -> K = j        + 8*(lane>=16)
//   j in 8..15 -> K = (j-8)+16 + 8*(lane>=16)
__device__ __forceinline__ v16h load_frag(const _Float16* base, int rs, int cs) {
  int lane = threadIdx.x & 31;
  int i    = lane & 15;
  int hk   = (lane >> 4) * 8;
  v16h f;
#pragma unroll
  for (int j = 0; j < 8; ++j) {
    f[j]     = base[i * rs + (hk + j) * cs];
    f[j + 8] = base[i * rs + (hk + 16 + j) * cs];
  }
  return f;
}

// ---- small convert --------------------------------------------------------
__global__ void f32_to_f16_kernel(const float* __restrict__ src,
                                  _Float16* __restrict__ dst, int n) {
  int i = blockIdx.x * blockDim.x + threadIdx.x;
  if (i < n) dst[i] = (_Float16)src[i];
}

// ---- grouped conv1d (same pad), f32 accumulate, f16 out -------------------
// transposed=0: out[((g*16+b)*L + l)*128 + d], o-fastest threads (qh/kh)
// transposed=1: out[(b*384 + o)*L + l],        l-fastest threads (vhT)
__global__ void grouped_conv_kernel(const float* __restrict__ x,
                                    const float* __restrict__ w,
                                    const float* __restrict__ bias,
                                    _Float16* __restrict__ out,
                                    int L, int transposed) {
  int idx = blockIdx.x * blockDim.x + threadIdx.x;
  int total = B_ * L * DMODEL_;
  if (idx >= total) return;
  int o, l, b;
  if (transposed) {
    l = idx % L;
    o = (idx / L) % DMODEL_;
    b = idx / (L * DMODEL_);
  } else {
    o = idx % DMODEL_;
    l = (idx / DMODEL_) % L;
    b = idx / (DMODEL_ * L);
  }
  int g = o >> 7;        // group / head
  int d = o & 127;
  const float* wo = w + (size_t)o * (128 * KW_);
  __builtin_prefetch(wo, 0, 0);                               // global_prefetch
  __builtin_prefetch(x + ((size_t)b * L + l) * DMODEL_ + g * 128, 0, 0);
  float acc = bias[o];
  for (int t = 0; t < KW_; ++t) {
    int ls = l + t - PADW_;
    if ((unsigned)ls >= (unsigned)L) continue;
    const float* xr = x + ((size_t)b * L + ls) * DMODEL_ + g * 128;
    const float* wt = wo + t;                 // w[o][c][t], c-stride = 13
#pragma unroll 8
    for (int c = 0; c < 128; ++c) acc += xr[c] * wt[c * KW_];
  }
  size_t oi = transposed ? (((size_t)b * DMODEL_ + o) * L + l)
                         : (((size_t)(g * B_ + b) * L + l) * 128 + d);
  out[oi] = (_Float16)acc;
}

// ---- energy GEMM: e[hb,q,k] = (1/sqrt(128)) * sum_d qh[hb,q,d]*kh[hb,k,d] --
// 48*(32 q-tiles)*(64 k-tiles) = 98304 wave tiles, 8 waves/block.
__global__ void energy_gemm_kernel(const _Float16* __restrict__ qh,
                                   const _Float16* __restrict__ kh,
                                   float* __restrict__ e) {
  int wave = blockIdx.x * (blockDim.x >> 5) + (threadIdx.x >> 5);
  int kt = wave & 63;
  int qt = (wave >> 6) & 31;
  int hb = wave >> 11;                 // 0..47  (h*16+b)
  const _Float16* A  = qh + (size_t)hb * QLEN_ * 128 + (size_t)qt * 16 * 128;
  const _Float16* Bm = kh + (size_t)hb * KLEN_ * 128 + (size_t)kt * 16 * 128;
  v8f c = {};
#pragma unroll
  for (int k0 = 0; k0 < 128; k0 += 32) {
    v16h a = load_frag(A  + k0, 128, 1);   // A(m,kk)=qh[q0+m][k0+kk]
    v16h b = load_frag(Bm + k0, 128, 1);   // B(kk,n)=kh[n0+n][k0+kk]
    c = __builtin_amdgcn_wmma_f32_16x16x32_f16(false, a, false, b,
                                               (short)0, c, false, false);
  }
  float* out = e + (size_t)hb * QLEN_ * KLEN_ + (size_t)qt * 16 * KLEN_ + kt * 16;
  int lane = threadIdx.x & 31;
  int n = lane & 15, mb = (lane >> 4) * 8;
  const float scale = 0.0883883476483184f;   // 1/sqrt(128)
#pragma unroll
  for (int r = 0; r < 8; ++r) out[(mb + r) * KLEN_ + n] = c[r] * scale;
}

// ---- row softmax over k (in place), one block per (hb,q) row --------------
__global__ void softmax_rows_kernel(float* __restrict__ e) {
  float* row = e + (size_t)blockIdx.x * KLEN_;
  __shared__ float red[256];
  int t = threadIdx.x;
  float m = -1e30f;
  for (int i = t; i < KLEN_; i += 256) m = fmaxf(m, row[i]);
  red[t] = m; __syncthreads();
  for (int s = 128; s > 0; s >>= 1) {
    if (t < s) red[t] = fmaxf(red[t], red[t + s]);
    __syncthreads();
  }
  m = red[0]; __syncthreads();
  float sum = 0.f;
  for (int i = t; i < KLEN_; i += 256) {
    float v = __expf(row[i] - m);
    row[i] = v; sum += v;
  }
  red[t] = sum; __syncthreads();
  for (int s = 128; s > 0; s >>= 1) {
    if (t < s) red[t] += red[t + s];
    __syncthreads();
  }
  float inv = 1.0f / red[0];
  for (int i = t; i < KLEN_; i += 256) row[i] *= inv;
}

// ---- fr[b,h] = (sum_k max_q alpha[hb,q,k]) / mel_len[b] -------------------
__global__ void fr_kernel(const float* __restrict__ alpha,
                          const int* __restrict__ mel_len,
                          float* __restrict__ fr) {
  int blk = blockIdx.x;              // = h*16 + b
  int h = blk >> 4, b = blk & 15;
  const float* base = alpha + (size_t)blk * QLEN_ * KLEN_;
  int k = threadIdx.x;               // 0..1023
  float m = 0.f;                     // softmax outputs > 0
  for (int q = 0; q < QLEN_; ++q) m = fmaxf(m, base[(size_t)q * KLEN_ + k]);
  __shared__ float red[1024];
  red[k] = m; __syncthreads();
  for (int s = 512; s > 0; s >>= 1) {
    if (k < s) red[k] += red[k + s];
    __syncthreads();
  }
  if (k == 0) fr[b * NHEAD_ + h] = red[0] / (float)mel_len[b];
}

// ---- per-batch argmax head + fr_max ---------------------------------------
__global__ void headsel_kernel(const float* __restrict__ fr,
                               int* __restrict__ hidx,
                               float* __restrict__ fr_max_out) {
  if (threadIdx.x == 0) {
    float acc = 0.f;
    for (int b = 0; b < B_; ++b) {
      int best = 0; float bv = fr[b * NHEAD_];
      for (int h = 1; h < NHEAD_; ++h) {
        float v = fr[b * NHEAD_ + h];
        if (v > bv) { bv = v; best = h; }
      }
      hidx[b] = best;
      acc += bv;
    }
    *fr_max_out = acc / (float)B_;
  }
}

// ---- select focused head, write alpha_fc output, write sigmoid(alpha)
//      transposed to [b,k,q] for the coalesced scan ------------------------
__global__ void select_transpose_kernel(const float* __restrict__ alpha,
                                        const int* __restrict__ hidx,
                                        float* __restrict__ alpha_fc_out,
                                        float* __restrict__ p_t) {
  __shared__ float tile[32][33];
  int blk = blockIdx.x;                    // b * 512 + qt * 32 + kt
  int b  = blk / (16 * 32);
  int rem = blk % (16 * 32);
  int q0 = (rem / 32) * 32;
  int k0 = (rem % 32) * 32;
  int hsel = hidx[b];
  const float* src = alpha + ((size_t)(hsel * B_ + b) * QLEN_) * KLEN_;
#pragma unroll
  for (int it = 0; it < 4; ++it) {
    int idx = threadIdx.x + 256 * it;
    int qi = idx >> 5, ki = idx & 31;
    float s = src[(size_t)(q0 + qi) * KLEN_ + k0 + ki];
    tile[qi][ki] = s;
    alpha_fc_out[((size_t)b * QLEN_ + q0 + qi) * KLEN_ + k0 + ki] = s;
  }
  __syncthreads();
#pragma unroll
  for (int it = 0; it < 4; ++it) {
    int idx = threadIdx.x + 256 * it;
    int kj = idx >> 5, qj = idx & 31;
    float s = tile[qj][kj];
    p_t[((size_t)b * KLEN_ + k0 + kj) * QLEN_ + q0 + qj] =
        1.f / (1.f + __expf(-s));
  }
}

// ---- stepwise monotonic expectation scan ----------------------------------
// 16 blocks x 512 threads; state aw[q] in registers, neighbor shift via LDS.
// Buffers 32 k-steps in an LDS tile, then flushes transposed so alpha_me is
// f16 row-major [b][q][k] (64B-contiguous per-thread stores, b128 A-loads).
__global__ void scan_kernel(const float* __restrict__ p_t,
                            _Float16* __restrict__ amh) {
  int b = blockIdx.x;
  int q = threadIdx.x;
  __shared__ float    mv[QLEN_];
  __shared__ _Float16 tk[32][QLEN_];          // 32 KB
  float aw = (q == 0) ? 1.f : 0.f;
  const float* pb = p_t + (size_t)b * KLEN_ * QLEN_;
  _Float16* ab = amh + (size_t)b * QLEN_ * KLEN_;   // [q][k]
  for (int k = 0; k < KLEN_; ++k) {
    float p = pb[(size_t)k * QLEN_ + q];
    mv[q] = aw * (1.f - p);
    __syncthreads();
    aw = aw * p + ((q > 0) ? mv[q - 1] : 0.f);
    tk[k & 31][q] = (_Float16)aw;
    __syncthreads();
    if ((k & 31) == 31) {                     // flush transposed chunk
      _Float16* dst = ab + (size_t)q * KLEN_ + (k - 31);
#pragma unroll
      for (int kk = 0; kk < 32; ++kk) dst[kk] = tk[kk][q];
    }
  }
}

// ---- cv GEMM: cv[b,q,d] = sum_k alpha_me[b,q,k] * v[b,k,d] ----------------
// A = alpha_me [b][q][k] row-major, B = vT [b][d][k] (both k-contiguous).
// 16*(32 q-tiles)*(24 d-tiles) = 12288 wave tiles.
__global__ void cv_gemm_kernel(const _Float16* __restrict__ amh,
                               const _Float16* __restrict__ vhT,
                               _Float16* __restrict__ cvh) {
  int wave = blockIdx.x * (blockDim.x >> 5) + (threadIdx.x >> 5);
  int dt = wave % 24;
  int qt = (wave / 24) % 32;
  int b  = wave / (24 * 32);
  const _Float16* Ab = amh + ((size_t)b * QLEN_ + qt * 16) * KLEN_;
  const _Float16* Bb = vhT + ((size_t)b * DMODEL_ + dt * 16) * KLEN_;
  v8f c = {};
  for (int k0 = 0; k0 < KLEN_; k0 += 32) {
    v16h a  = load_frag(Ab + k0, KLEN_, 1);   // A(m,kk)=alpha[q0+m][k0+kk]
    v16h bb = load_frag(Bb + k0, KLEN_, 1);   // B(kk,n)=vT[d0+n][k0+kk]
    c = __builtin_amdgcn_wmma_f32_16x16x32_f16(false, a, false, bb,
                                               (short)0, c, false, false);
  }
  _Float16* out = cvh + ((size_t)b * QLEN_ + qt * 16) * DMODEL_ + dt * 16;
  int lane = threadIdx.x & 31;
  int n = lane & 15, mb = (lane >> 4) * 8;
#pragma unroll
  for (int r = 0; r < 8; ++r) out[(mb + r) * DMODEL_ + n] = (_Float16)c[r];
}

// ---- final projection: out = cv @ w_last^T + b_last -----------------------
// (512 m-tiles)*(24 n-tiles) = 12288 wave tiles.
__global__ void final_gemm_kernel(const _Float16* __restrict__ cvh,
                                  const _Float16* __restrict__ wlh,
                                  const float* __restrict__ b_last,
                                  float* __restrict__ outp) {
  int wave = blockIdx.x * (blockDim.x >> 5) + (threadIdx.x >> 5);
  int nt = wave % 24;
  int mt = wave / 24;
  const _Float16* Ab = cvh + (size_t)mt * 16 * DMODEL_;
  const _Float16* Bb = wlh + (size_t)nt * 16 * DMODEL_;   // w_last[n][j]
  v8f c = {};
#pragma unroll 2
  for (int k0 = 0; k0 < DMODEL_; k0 += 32) {
    v16h a  = load_frag(Ab + k0, DMODEL_, 1);   // A(m,kk)=cv[m][j]
    v16h bb = load_frag(Bb + k0, DMODEL_, 1);   // B(kk,n)=w_last[n][j]
    c = __builtin_amdgcn_wmma_f32_16x16x32_f16(false, a, false, bb,
                                               (short)0, c, false, false);
  }
  float* out = outp + (size_t)mt * 16 * DMODEL_ + nt * 16;
  int lane = threadIdx.x & 31;
  int n = lane & 15, mb = (lane >> 4) * 8;
#pragma unroll
  for (int r = 0; r < 8; ++r)
    out[(mb + r) * DMODEL_ + n] = c[r] + b_last[nt * 16 + n];
}

// ---------------------------------------------------------------------------
extern "C" void kernel_launch(void* const* d_in, const int* in_sizes, int n_in,
                              void* d_out, int out_size, void* d_ws, size_t ws_size,
                              hipStream_t stream) {
  const float* q      = (const float*)d_in[0];
  const float* k      = (const float*)d_in[1];
  const float* v      = (const float*)d_in[2];
  const int*   mel    = (const int*)d_in[3];
  const float* w_qs   = (const float*)d_in[4];
  const float* b_qs   = (const float*)d_in[5];
  const float* w_ks   = (const float*)d_in[6];
  const float* b_ks   = (const float*)d_in[7];
  const float* w_vs   = (const float*)d_in[8];
  const float* b_vs   = (const float*)d_in[9];
  const float* w_last = (const float*)d_in[10];
  const float* b_last = (const float*)d_in[11];

  float* out_cv    = (float*)d_out;                                   // [16,512,384]
  float* out_alpha = (float*)d_out + (size_t)B_ * QLEN_ * DMODEL_;    // [16,512,1024]
  float* out_frmax = (float*)d_out + (size_t)B_ * QLEN_ * DMODEL_
                                   + (size_t)B_ * QLEN_ * KLEN_;      // scalar

  // workspace layout (256B aligned)
  char* ws = (char*)d_ws;
  size_t off = 0;
  auto take = [&](size_t bytes) -> char* {
    char* p = ws + off;
    off = (off + bytes + 255) & ~(size_t)255;
    return p;
  };
  _Float16* wlh  = (_Float16*)take((size_t)DMODEL_ * DMODEL_ * 2);
  _Float16* qh16 = (_Float16*)take((size_t)NHEAD_ * B_ * QLEN_ * 128 * 2);
  _Float16* kh16 = (_Float16*)take((size_t)NHEAD_ * B_ * KLEN_ * 128 * 2);
  _Float16* vhT  = (_Float16*)take((size_t)B_ * DMODEL_ * KLEN_ * 2);
  float*    e    = (float*)   take((size_t)NHEAD_ * B_ * QLEN_ * KLEN_ * 4);
  float*    fr   = (float*)   take((size_t)B_ * NHEAD_ * 4);
  int*      hidx = (int*)     take((size_t)B_ * 4);
  float*    p_t  = (float*)   take((size_t)B_ * KLEN_ * QLEN_ * 4);
  _Float16* amh  = (_Float16*)take((size_t)B_ * QLEN_ * KLEN_ * 2);
  _Float16* cvh  = (_Float16*)take((size_t)B_ * QLEN_ * DMODEL_ * 2);
  (void)ws_size; (void)in_sizes; (void)n_in; (void)out_size;

  // 1) w_last -> f16
  f32_to_f16_kernel<<<(DMODEL_ * DMODEL_ + 255) / 256, 256, 0, stream>>>(
      w_last, wlh, DMODEL_ * DMODEL_);

  // 2) grouped convs
  grouped_conv_kernel<<<(B_ * QLEN_ * DMODEL_ + 255) / 256, 256, 0, stream>>>(
      q, w_qs, b_qs, qh16, QLEN_, 0);
  grouped_conv_kernel<<<(B_ * KLEN_ * DMODEL_ + 255) / 256, 256, 0, stream>>>(
      k, w_ks, b_ks, kh16, KLEN_, 0);
  grouped_conv_kernel<<<(B_ * KLEN_ * DMODEL_ + 255) / 256, 256, 0, stream>>>(
      v, w_vs, b_vs, vhT, KLEN_, 1);

  // 3) energy GEMM (WMMA): 98304 wave tiles / 8 waves per block
  energy_gemm_kernel<<<12288, 256, 0, stream>>>(qh16, kh16, e);

  // 4) softmax over k (in place -> alpha_cv)
  softmax_rows_kernel<<<NHEAD_ * B_ * QLEN_, 256, 0, stream>>>(e);

  // 5) fr + head selection
  fr_kernel<<<NHEAD_ * B_, 1024, 0, stream>>>(e, mel, fr);
  headsel_kernel<<<1, 32, 0, stream>>>(fr, hidx, out_frmax);

  // 6) select focused head, emit alpha_fc + transposed sigmoid probs
  select_transpose_kernel<<<B_ * 16 * 32, 256, 0, stream>>>(e, hidx,
                                                            out_alpha, p_t);

  // 7) monotonic expectation scan (LDS tile-transposed f16 output)
  scan_kernel<<<B_, QLEN_, 0, stream>>>(p_t, amh);

  // 8) context GEMM (WMMA): 12288 wave tiles / 8
  cv_gemm_kernel<<<1536, 256, 0, stream>>>(amh, vhT, cvh);

  // 9) final projection GEMM (WMMA) + bias
  final_gemm_kernel<<<1536, 256, 0, stream>>>(cvh, wlh, b_last, out_cv);
}
